// TSPActorNSA_2619930050646
// MI455X (gfx1250) — matrix operand order repaired and checked by reading, not compiled, not run on hardware
//
#include <hip/hip_runtime.h>
#include <math.h>

typedef __attribute__((ext_vector_type(8)))  _Float16 v8h;
typedef __attribute__((ext_vector_type(16))) _Float16 v16h;
typedef __attribute__((ext_vector_type(8)))  float    v8f;
typedef __attribute__((ext_vector_type(4)))  int      v4i;

#define NN 1024
#define PP 1024
#define EE 128
#define NW 2          // waves per block
#define BD (NW * 32)  // blockDim = 64

// f16 weight image in d_ws (halfs):
//   [0)      w1a_pad [128][32]   (4096)   col 7  = b1a (bias folded, feature=1.0)
//   [4096)   w1b     [128][128]  (16384)
//   [20480)  w2a_pad [128][32]   (4096)   col 13 = b2a
//   [24576)  w2b     [128][128]  (16384)
#define WS_PHASE_HALFS 20480
#define WS_PHASE_BYTES (WS_PHASE_HALFS * 2)

#if defined(__AMDGCN__) && __has_builtin(__builtin_amdgcn_global_load_async_to_lds_b128) && \
    __has_builtin(__builtin_amdgcn_s_wait_asynccnt)
#define USE_ASYNC_LDS 1
#else
#define USE_ASYNC_LDS 0
#endif

// single-instruction ReLU: v_med3_f32(x, 0, +inf)
__device__ __forceinline__ float relu(float x) {
#if defined(__AMDGCN__) && __has_builtin(__builtin_amdgcn_fmed3f)
    return __builtin_amdgcn_fmed3f(x, 0.0f, __builtin_huge_valf());
#else
    return fmaxf(x, 0.0f);
#endif
}

// ---- A-fragment (16x32 f16) = weight tile, from row-major W[out][kdim] ----
// lane<16: K = kb+0..15; lane>=16: K = kb+16..31; M = lane&15
__device__ __forceinline__ v16h load_aw(const _Float16* w, int mt, int kdim, int kb, int lane) {
    int row = mt * 16 + (lane & 15);
    int k0  = kb + ((lane & 16) ? 16 : 0);
    v8h lo = *(const v8h*)(w + row * kdim + k0);
    v8h hi = *(const v8h*)(w + row * kdim + k0 + 8);
    v16h r;
#pragma unroll
    for (int i = 0; i < 8; ++i) { r[i] = lo[i]; r[i + 8] = hi[i]; }
    return r;
}

// ---- B-fragment (32x16 f16) = activation tile, from row-major act[token][stride] ----
// lane<16: K = kb+0..15 of token (lane&15); lane>=16: K = kb+16..31
__device__ __forceinline__ v16h load_bact(const _Float16* act, int stride, int kb, int lane) {
    return *(const v16h*)(act + (lane & 15) * stride + kb + ((lane & 16) ? 16 : 0));
}

// Dual-tile MLP phase, transposed orientation: D = W x act^T.
__device__ __forceinline__ void mlp_phase(
    const float* __restrict__ S, const int* __restrict__ action,
    int n, int pblk, int lane, int wave,
    const _Float16* s_wa, const _Float16* s_wb,
    _Float16* featA, _Float16* featB, _Float16* hA, _Float16* hB,
    const float* __restrict__ Bb, const float* __restrict__ Wc,
    float* __restrict__ outRow, bool second)
{
    int ncol = lane & 15;                 // token within tile
    int m0   = (lane & 16) ? 8 : 0;       // feat-out half within 16-row tile

    // c2 "base" features (row-uniform), computed by every lane
    float bf[6] = {0.f, 0.f, 0.f, 0.f, 0.f, 0.f};
    if (second) {
        int c1 = action[2 * n];
        int cm = (c1 + PP - 1) & (PP - 1);
        int cp = (c1 + 1) & (PP - 1);
        int x0 = (int)S[4 * c1], x1 = (int)S[4 * cm], x2 = (int)S[4 * cp];
        bf[0] = S[4 * x0 + 1]; bf[1] = S[4 * x0 + 2];
        bf[2] = S[4 * x1 + 1]; bf[3] = S[4 * x1 + 2];
        bf[4] = S[4 * x2 + 1]; bf[5] = S[4 * x2 + 2];
    }

    for (int pass = 0; pass < 2; ++pass) {
        int p0A = pblk + wave * 64 + pass * 32;
        int p0B = p0A + 16;

        // ---- features: low lanes build tile A token, high lanes tile B ----
        {
            int tok = ncol;
            int p   = ((lane < 16) ? p0A : p0B) + tok;
            _Float16* fb = (lane < 16) ? featA : featB;
            __align__(16) _Float16 row[32];
#pragma unroll
            for (int k = 0; k < 32; ++k) row[k] = (_Float16)0.0f;
            int pm = (p + PP - 1) & (PP - 1);
            int pn = (p + 1) & (PP - 1);
            int xi  = (int)S[4 * p];
            int xim = (int)S[4 * pm];
            int xip = (int)S[4 * pn];
            int off = 0;
            if (second) {
#pragma unroll
                for (int k = 0; k < 6; ++k) row[k] = (_Float16)bf[k];
                off = 6;
            }
            row[off + 0] = (_Float16)S[4 * xi + 1];
            row[off + 1] = (_Float16)S[4 * xi + 2];
            row[off + 2] = (_Float16)S[4 * xim + 1];
            row[off + 3] = (_Float16)S[4 * xim + 2];
            row[off + 4] = (_Float16)S[4 * xip + 1];
            row[off + 5] = (_Float16)S[4 * xip + 2];
            row[off + 6] = (_Float16)S[4 * p + 3];
            row[off + 7] = (_Float16)1.0f;   // bias column (weights carry the bias)
            v8h* dst = (v8h*)(fb + tok * 32);
            const v8h* src = (const v8h*)row;
            dst[0] = src[0]; dst[1] = src[1]; dst[2] = src[2]; dst[3] = src[3];
        }
        // feat/h are per-wave private; LDS is in-order per wave -> no barrier.

        // ---- layer a: D[feat_out][token], K=32 padded, bias folded, C = inline 0 ----
        v16h bA = load_bact(featA, 32, 0, lane);
        v16h bB = load_bact(featB, 32, 0, lane);
#pragma unroll
        for (int mt = 0; mt < 8; ++mt) {
            v16h aw = load_aw(s_wa, mt, 32, 0, lane);
            v8f cA = {}, cB = {};
            cA = __builtin_amdgcn_wmma_f32_16x16x32_f16(false, aw, false, bA, (short)0, cA, false, false);
            cB = __builtin_amdgcn_wmma_f32_16x16x32_f16(false, aw, false, bB, (short)0, cB, false, false);
            // lane's 8 outputs = 8 consecutive features of its token -> packed b128 store
            v8h hvA, hvB;
#pragma unroll
            for (int r = 0; r < 8; ++r) {
                hvA[r] = (_Float16)relu(cA[r]);
                hvB[r] = (_Float16)relu(cB[r]);
            }
            *(v8h*)(hA + ncol * EE + mt * 16 + m0) = hvA;
            *(v8h*)(hB + ncol * EE + mt * 16 + m0) = hvB;
        }

        // ---- layer b (K=128) + fused layer c dot ----
        v16h xA[4], xB[4];
#pragma unroll
        for (int kc = 0; kc < 4; ++kc) {
            xA[kc] = load_bact(hA, EE, kc * 32, lane);
            xB[kc] = load_bact(hB, EE, kc * 32, lane);
        }
        float accA = 0.0f, accB = 0.0f;
#pragma unroll
        for (int mt = 0; mt < 8; ++mt) {
            v8f cA = {}, cB = {};
#pragma unroll
            for (int kc = 0; kc < 4; ++kc) {
                v16h aw = load_aw(s_wb, mt, EE, kc * 32, lane);
                cA = __builtin_amdgcn_wmma_f32_16x16x32_f16(false, aw, false, xA[kc], (short)0, cA, false, false);
                cB = __builtin_amdgcn_wmma_f32_16x16x32_f16(false, aw, false, xB[kc], (short)0, cB, false, false);
            }
            // per-lane feat-out slice: mt*16 + m0 + r  (aligned float4 loads)
            float4 bb0 = *(const float4*)(Bb + mt * 16 + m0);
            float4 bb1 = *(const float4*)(Bb + mt * 16 + m0 + 4);
            float4 wc0 = *(const float4*)(Wc + mt * 16 + m0);
            float4 wc1 = *(const float4*)(Wc + mt * 16 + m0 + 4);
            float bbv[8] = {bb0.x, bb0.y, bb0.z, bb0.w, bb1.x, bb1.y, bb1.z, bb1.w};
            float wcv[8] = {wc0.x, wc0.y, wc0.z, wc0.w, wc1.x, wc1.y, wc1.z, wc1.w};
#pragma unroll
            for (int r = 0; r < 8; ++r) {
                accA += wcv[r] * relu(cA[r] + bbv[r]);
                accB += wcv[r] * relu(cB[r] + bbv[r]);
            }
        }
        // lane l and lane l+16 hold complementary feat-halves of the same token
        accA += __shfl_xor(accA, 16, 32);
        accB += __shfl_xor(accB, 16, 32);
        if (lane < 16) {
            outRow[p0A + ncol] = accA;
            outRow[p0B + ncol] = accB;
        }
    }
}

// ---- one-shot fp32 -> f16 weight image into d_ws (layer-a bias folded) ----
__global__ __launch_bounds__(256) void tsp_prep_weights(
    const float* __restrict__ W1a, const float* __restrict__ b1a,
    const float* __restrict__ W1b,
    const float* __restrict__ W2a, const float* __restrict__ b2a,
    const float* __restrict__ W2b,
    _Float16* __restrict__ ws)
{
    int i = blockIdx.x * 256 + threadIdx.x;   // 0..40959
    if (i < 4096) {
        int r = i >> 5, k = i & 31;
        float v = (k < 7) ? W1a[r * 7 + k] : (k == 7 ? b1a[r] : 0.0f);
        ws[i] = (_Float16)v;
    } else if (i < 20480) {
        ws[i] = (_Float16)W1b[i - 4096];
    } else if (i < 24576) {
        int j = i - 20480, r = j >> 5, k = j & 31;
        float v = (k < 13) ? W2a[r * 13 + k] : (k == 13 ? b2a[r] : 0.0f);
        ws[i] = (_Float16)v;
    } else if (i < 40960) {
        ws[i] = (_Float16)W2b[i - 24576];
    }
}

// waves_per_eu >= 4 caps VGPRs at 256: keeps allocation in the direct window,
// eliminating the s_set_vgpr_msb toggling seen with an uncapped budget.
__global__ __launch_bounds__(BD, 4) void tsp_mlp_kernel(
    const float* __restrict__ state, const int* __restrict__ action,
    const float* __restrict__ b1b, const float* __restrict__ W1c,
    const float* __restrict__ b2b, const float* __restrict__ W2c,
    const _Float16* __restrict__ ws, float* __restrict__ out)
{
    __shared__ __align__(32) _Float16 s_w[WS_PHASE_HALFS];     // 40 KB: wa | wb
    __shared__ __align__(32) _Float16 s_feat[NW][2][16 * 32];  // 4 KB
    __shared__ __align__(32) _Float16 s_h[NW][2][16 * EE];     // 16 KB

    int n    = blockIdx.x >> 3;                 // row (0..1023)
    int wave = threadIdx.x >> 5;                // 0..1
    int lane = threadIdx.x & 31;
    int pblk = (blockIdx.x & 7) << 7;           // 128 tokens per block
    int tid  = threadIdx.x;

    const float* S = state + (size_t)n * PP * 4;
    _Float16* s_wa = s_w;
    _Float16* s_wb = s_w + 4096;

    for (int ph = 0; ph < 2; ++ph) {
        __syncthreads();   // all waves done reading previous phase's weights
        const _Float16* src = ws + (size_t)ph * WS_PHASE_HALFS;
#if USE_ASYNC_LDS
        {
            // builtin expects (v4i AS1*, v4i AS3*, imm offset, imm cpol)
            v4i* gsrc_gen = (v4i*)const_cast<_Float16*>(src);
            __attribute__((address_space(1))) v4i* g =
                (__attribute__((address_space(1))) v4i*)gsrc_gen;
            __attribute__((address_space(3))) v4i* l =
                (__attribute__((address_space(3))) v4i*)(v4i*)s_w;
            for (int i = tid; i < WS_PHASE_BYTES / 16; i += BD)
                __builtin_amdgcn_global_load_async_to_lds_b128(g + i, l + i, 0, 0);
            __builtin_amdgcn_s_wait_asynccnt(0);
        }
#else
        for (int i = tid; i < WS_PHASE_HALFS / 8; i += BD)
            ((v8h*)s_w)[i] = ((const v8h*)src)[i];
#endif
        __syncthreads();

        if (ph == 0) {
            mlp_phase(S, action, n, pblk, lane, wave, s_wa, s_wb,
                      s_feat[wave][0], s_feat[wave][1], s_h[wave][0], s_h[wave][1],
                      b1b, W1c, out + (size_t)n * PP, false);
        } else {
            mlp_phase(S, action, n, pblk, lane, wave, s_wa, s_wb,
                      s_feat[wave][0], s_feat[wave][1], s_h[wave][0], s_h[wave][1],
                      b2b, W2c, out + (size_t)NN * PP + (size_t)n * PP, true);
        }
    }
}

// In-place row-wise log_softmax over P=1024; second half masks c1, c1+-1.
__global__ __launch_bounds__(256) void tsp_logsoftmax_kernel(
    float* __restrict__ out, const int* __restrict__ action)
{
    int b = blockIdx.x;             // 0..2047
    bool second = b >= NN;
    int n = second ? b - NN : b;
    float* row = out + (second ? (size_t)NN * PP : 0) + (size_t)n * PP;
    int tid = threadIdx.x;

    int c1 = -1, cm = -1, cp = -1;
    if (second) {
        c1 = action[2 * n];
        cm = (c1 + PP - 1) & (PP - 1);
        cp = (c1 + 1) & (PP - 1);
    }
    float v[4];
    float mx = -INFINITY;
#pragma unroll
    for (int i = 0; i < 4; ++i) {
        int idx = i * 256 + tid;
        float x = row[idx];
        if (second && (idx == c1 || idx == cm || idx == cp)) x = -INFINITY;
        v[i] = x;
        mx = fmaxf(mx, x);
    }
    __shared__ float red[256];
    red[tid] = mx;
    __syncthreads();
    for (int s = 128; s > 0; s >>= 1) {
        if (tid < s) red[tid] = fmaxf(red[tid], red[tid + s]);
        __syncthreads();
    }
    mx = red[0];
    __syncthreads();
    float sm = 0.0f;
#pragma unroll
    for (int i = 0; i < 4; ++i) sm += expf(v[i] - mx);
    red[tid] = sm;
    __syncthreads();
    for (int s = 128; s > 0; s >>= 1) {
        if (tid < s) red[tid] += red[tid + s];
        __syncthreads();
    }
    float lse = mx + logf(red[0]);
#pragma unroll
    for (int i = 0; i < 4; ++i) row[i * 256 + tid] = v[i] - lse;
}

extern "C" void kernel_launch(void* const* d_in, const int* in_sizes, int n_in,
                              void* d_out, int out_size, void* d_ws, size_t ws_size,
                              hipStream_t stream) {
    const float* state = (const float*)d_in[0];
    const int*   action = (const int*)d_in[1];
    const float* W1a = (const float*)d_in[2];
    const float* b1a = (const float*)d_in[3];
    const float* W1b = (const float*)d_in[4];
    const float* b1b = (const float*)d_in[5];
    const float* W1c = (const float*)d_in[6];
    const float* W2a = (const float*)d_in[7];
    const float* b2a = (const float*)d_in[8];
    const float* W2b = (const float*)d_in[9];
    const float* b2b = (const float*)d_in[10];
    const float* W2c = (const float*)d_in[11];
    float* out = (float*)d_out;
    _Float16* wsw = (_Float16*)d_ws;

    tsp_prep_weights<<<160, 256, 0, stream>>>(W1a, b1a, W1b, W2a, b2a, W2b, wsw);
    // 1024 rows x 8 blocks/row; block = 2 waves x 2 passes x 2 tiles = 128 tokens
    tsp_mlp_kernel<<<NN * 8, BD, 0, stream>>>(
        state, action, b1b, W1c, b2b, W2c, wsw, out);
    tsp_logsoftmax_kernel<<<2 * NN, 256, 0, stream>>>(out, action);
    (void)in_sizes; (void)n_in; (void)out_size; (void)ws_size;
}
